// CausalSelfAttention_46102178955902
// MI455X (gfx1250) — matrix operand order repaired and checked by reading, not compiled, not run on hardware
//
#include <hip/hip_runtime.h>
#include <hip/hip_bf16.h>
#include <math.h>

// ---------------------------------------------------------------------------
// CDNA5 (gfx1250) causal self-attention: bf16 WMMA everywhere, fp32 accum,
// async LDS staging (GLOBAL_LOAD_ASYNC_TO_LDS_B128 / S_WAIT_ASYNCCNT) for K/V.
// ---------------------------------------------------------------------------

typedef __bf16 bf16_t;
typedef __attribute__((ext_vector_type(16))) __bf16 v16bf;
typedef __attribute__((ext_vector_type(8)))  __bf16 v8bf;
typedef __attribute__((ext_vector_type(8)))  float  v8f;

#define TSEQ  4096
#define NBAT  2
#define NHEAD 12
#define HDIM  64
#define CEMB  768
#define MTOT  (NBAT*TSEQ)   // 8192 rows

static __device__ __forceinline__ v16bf cat16(v8bf lo, v8bf hi){
  return __builtin_shufflevector(lo, hi, 0,1,2,3,4,5,6,7,8,9,10,11,12,13,14,15);
}
// B-fragment: 16 contiguous bf16 (k = 16*half + e), 32B
static __device__ __forceinline__ v16bf ldB(const bf16_t* p){
  return cat16(*(const v8bf*)p, *(const v8bf*)(p + 8));
}
// A-fragment: 16 contiguous bf16 (k = 16*half + e), 32B
static __device__ __forceinline__ v16bf ldA(const bf16_t* p){
  return cat16(*(const v8bf*)p, *(const v8bf*)(p + 8));
}
static __device__ __forceinline__ v8f wmma_bf(v16bf a, v16bf b, v8f c){
  return __builtin_amdgcn_wmma_f32_16x16x32_bf16(false, a, false, b, (short)0, c,
                                                 false, false);
}
// row reductions across the 16 lanes that share a C-fragment row (masks<16
// never cross the half boundary)
static __device__ __forceinline__ float rmax16(float v){
  v = fmaxf(v, __shfl_xor(v, 1)); v = fmaxf(v, __shfl_xor(v, 2));
  v = fmaxf(v, __shfl_xor(v, 4)); v = fmaxf(v, __shfl_xor(v, 8));
  return v;
}
static __device__ __forceinline__ float rsum16(float v){
  v += __shfl_xor(v, 1); v += __shfl_xor(v, 2);
  v += __shfl_xor(v, 4); v += __shfl_xor(v, 8);
  return v;
}

// Async DMA of one 32-column KV tile into LDS (executed by all 128 threads).
//  K tile: 32 rows(t) x 64(d) bf16 = 4KB, contiguous in global.
//  V tile: 64 rows(d) x 32(t)  bf16 = 4KB, global row stride TSEQ.
static __device__ __forceinline__ void stage_kv(const bf16_t* __restrict__ Kh,
                                                const bf16_t* __restrict__ Vh,
                                                int kv, bf16_t* ksh, bf16_t* vsh,
                                                int tid)
{
  #pragma unroll
  for (int i = 0; i < 2; i++){
    const int c = tid + i*128;                       // 16B chunk id, 0..255
    unsigned kdst = (unsigned)(uintptr_t)ksh + (unsigned)c*16;
    unsigned long long ksrc =
        (unsigned long long)(uintptr_t)(Kh + (size_t)kv*HDIM + (size_t)c*8);
    asm volatile("global_load_async_to_lds_b128 %0, %1, off"
                 :: "v"(kdst), "v"(ksrc) : "memory");
  }
  #pragma unroll
  for (int i = 0; i < 2; i++){
    const int c = tid + i*128;
    const int d = c >> 2, ts = (c & 3) * 8;
    unsigned vdst = (unsigned)(uintptr_t)vsh + (unsigned)c*16;
    unsigned long long vsrc =
        (unsigned long long)(uintptr_t)(Vh + (size_t)d*TSEQ + kv + ts);
    asm volatile("global_load_async_to_lds_b128 %0, %1, off"
                 :: "v"(vdst), "v"(vsrc) : "memory");
  }
}

// ---------------------------------------------------------------------------
__global__ void cvt_bf16(const float* __restrict__ in, bf16_t* __restrict__ out, int n){
  int i = blockIdx.x * blockDim.x + threadIdx.x;
  int stride = gridDim.x * blockDim.x;
  for (; i < n; i += stride) out[i] = (bf16_t)in[i];
}

// ---------------------------------------------------------------------------
// QKV projection: C[m,n] = sum_k x[m,k] * W[n,k] + b[n]; each wave does 16x64.
// mode 0 -> Q (rope, [b][h][t][d]); 1 -> K (rope, same); 2 -> V^T ([b][h][d][t])
__global__ void gemm_qkv(const bf16_t* __restrict__ xb,
                         const bf16_t* __restrict__ wb,
                         const float*  __restrict__ bias,
                         bf16_t* __restrict__ dst,
                         int mode)
{
  const int lane  = threadIdx.x & 31;
  const int wv    = threadIdx.x >> 5;
  const int mtile = (blockIdx.x * 4 + wv) * 16;
  const int n0    = blockIdx.y * 64;
  const int nloc  = lane & 15, hf = lane >> 4;

  const bf16_t* arow = xb + (size_t)(mtile + nloc) * CEMB + hf * 16;
  const bf16_t* bp0  = wb + (size_t)(n0 +  0 + nloc) * CEMB + hf * 16;
  const bf16_t* bp1  = wb + (size_t)(n0 + 16 + nloc) * CEMB + hf * 16;
  const bf16_t* bp2  = wb + (size_t)(n0 + 32 + nloc) * CEMB + hf * 16;
  const bf16_t* bp3  = wb + (size_t)(n0 + 48 + nloc) * CEMB + hf * 16;

  v8f a0 = {}, a1 = {}, a2 = {}, a3 = {};
  #pragma unroll 4
  for (int kb = 0; kb < CEMB; kb += 32){
    v16bf af = ldA(arow + kb);
    a0 = wmma_bf(af, ldB(bp0 + kb), a0);
    a1 = wmma_bf(af, ldB(bp1 + kb), a1);
    a2 = wmma_bf(af, ldB(bp2 + kb), a2);
    a3 = wmma_bf(af, ldB(bp3 + kb), a3);
  }

  #pragma unroll
  for (int j = 0; j < 4; j++){
    v8f acc = (j==0)?a0:(j==1)?a1:(j==2)?a2:a3;
    const int n = n0 + j*16 + nloc;
    const float bv = bias[n];
    const int h = n >> 6, d = n & 63;
    // rope inv-freq: 10000^(-2i/64) = 2^(-(2i/64)*log2(1e4)), 2i = d&~1
    const float invf = exp2f(-13.2877124f * (float)(d & ~1) * (1.0f/64.0f));
    #pragma unroll
    for (int r = 0; r < 8; r++){
      const int mg = mtile + r + hf*8;
      const int t  = mg & (TSEQ - 1);
      const int bb = mg >> 12;                 // T = 4096
      float val = acc[r] + bv;
      if (mode == 2){
        // V transposed: [b][h][d][t]
        dst[(((size_t)bb*NHEAD + h)*HDIM + d)*TSEQ + t] = (bf16_t)val;
      } else {
        // RoPE: pair (even d, odd d) lives in adjacent lanes (n = lane&15)
        float partner = __shfl_xor(val, 1);
        float ang = (float)t * invf;
        float sn, cs; __sincosf(ang, &sn, &cs);
        float res = (d & 1) ? (partner*sn + val*cs) : (val*cs - partner*sn);
        dst[(((size_t)bb*NHEAD + h)*TSEQ + t)*HDIM + d] = (bf16_t)res;
      }
    }
  }
}

// ---------------------------------------------------------------------------
// Flash attention, block = 4 waves covering rows [R0, R0+64) of one (b,h).
// K/V tiles double-buffered in LDS via async DMA shared by all 4 waves.
__global__ void flash_attn(const bf16_t* __restrict__ Qb,
                           const bf16_t* __restrict__ Kb,
                           const bf16_t* __restrict__ Vtb,
                           bf16_t* __restrict__ Yb)
{
  __shared__ __align__(16) bf16_t Ksh[2][32*HDIM];   // [buf][t][d]   8 KB
  __shared__ __align__(16) bf16_t Vsh[2][HDIM*32];   // [buf][d][t]   8 KB
  __shared__ float Plds[4 * 512];                    // P relayout    8 KB

  const int tid  = threadIdx.x;
  const int lane = tid & 31;
  const int wv   = tid >> 5;
  const int h    = blockIdx.y;
  const int bb   = blockIdx.z;
  const int R0   = blockIdx.x * 64;
  const int tq0  = R0 + wv * 16;
  const int nloc = lane & 15, hf = lane >> 4;

  const bf16_t* Qh = Qb  + (((size_t)bb*NHEAD + h)*TSEQ)*HDIM;
  const bf16_t* Kh = Kb  + (((size_t)bb*NHEAD + h)*TSEQ)*HDIM;
  const bf16_t* Vh = Vtb + (((size_t)bb*NHEAD + h)*HDIM)*TSEQ;

  const v16bf aQ0 = ldA(Qh + (size_t)(tq0 + nloc)*HDIM +  0 + hf*16);
  const v16bf aQ1 = ldA(Qh + (size_t)(tq0 + nloc)*HDIM + 32 + hf*16);

  v8f o0 = {}, o1 = {}, o2 = {}, o3 = {};
  float mrow[8], lrow[8];
  #pragma unroll
  for (int r = 0; r < 8; r++){ mrow[r] = -INFINITY; lrow[r] = 0.f; }

  float* P = Plds + wv * 512;
  const float sc = 0.125f;                     // 1/sqrt(64)
  const int kvend = R0 + 64;                   // uniform across the block

  stage_kv(Kh, Vh, 0, Ksh[0], Vsh[0], tid);

  for (int kv = 0; kv < kvend; kv += 32){
    const int cur = (kv >> 5) & 1;
    // this wave's DMAs for buffer `cur` are complete...
    asm volatile("s_wait_asynccnt 0x0" ::: "memory");
    // ...and after the barrier, everyone's are; previous buffer's readers
    // are also past this point, so it is safe to restage it below.
    __syncthreads();
    if (kv + 32 < kvend)
      stage_kv(Kh, Vh, kv + 32, Ksh[cur ^ 1], Vsh[cur ^ 1], tid);  // overlaps compute

    if (kv <= tq0 + 15){
      const bf16_t* kt = Ksh[cur];
      const bf16_t* vt = Vsh[cur];
      v8f s0 = {}, s1 = {};
      {
        const bf16_t* kp0 = kt + (size_t)nloc*HDIM + hf*16;
        const bf16_t* kp1 = kt + (size_t)(16 + nloc)*HDIM + hf*16;
        s0 = wmma_bf(aQ0, ldB(kp0),      s0);
        s0 = wmma_bf(aQ1, ldB(kp0 + 32), s0);
        s1 = wmma_bf(aQ0, ldB(kp1),      s1);
        s1 = wmma_bf(aQ1, ldB(kp1 + 32), s1);
      }
      const int c0 = kv + nloc, c1 = kv + 16 + nloc;
      float p0[8], p1[8];
      #pragma unroll
      for (int r = 0; r < 8; r++){
        const int row = tq0 + r + hf*8;
        float v0 = s0[r]*sc; if (c0 > row) v0 = -INFINITY;
        float v1 = s1[r]*sc; if (c1 > row) v1 = -INFINITY;
        float mnew  = fmaxf(mrow[r], rmax16(fmaxf(v0, v1)));
        float alpha = __expf(mrow[r] - mnew);
        mrow[r] = mnew;
        v0 = __expf(v0 - mnew); v1 = __expf(v1 - mnew);
        lrow[r] = lrow[r]*alpha + rsum16(v0 + v1);
        o0[r] *= alpha; o1[r] *= alpha; o2[r] *= alpha; o3[r] *= alpha;
        p0[r] = v0; p1[r] = v1;
      }
      // C-layout -> LDS (row-major 16x32). Same-wave DS ops are in-order;
      // just stop the compiler from reordering across the relayout.
      #pragma unroll
      for (int r = 0; r < 8; r++){
        P[(r + hf*8)*32 + nloc]      = p0[r];
        P[(r + hf*8)*32 + 16 + nloc] = p1[r];
      }
      __asm__ volatile("" ::: "memory");
      v16bf ap;
      #pragma unroll
      for (int e = 0; e < 16; e++){
        const int k = hf*16 + e;   // A-frag k index
        ap[e] = (bf16_t)P[nloc*32 + k];
      }
      __asm__ volatile("" ::: "memory");
      o0 = wmma_bf(ap, ldB(vt + (size_t)( 0 + nloc)*32 + hf*16), o0);
      o1 = wmma_bf(ap, ldB(vt + (size_t)(16 + nloc)*32 + hf*16), o1);
      o2 = wmma_bf(ap, ldB(vt + (size_t)(32 + nloc)*32 + hf*16), o2);
      o3 = wmma_bf(ap, ldB(vt + (size_t)(48 + nloc)*32 + hf*16), o3);
    }
  }

  #pragma unroll
  for (int r = 0; r < 8; r++){
    const int t = tq0 + r + hf*8;
    const float inv = 1.0f / lrow[r];
    const size_t base = ((size_t)bb*TSEQ + t)*CEMB + h*HDIM;
    Yb[base +  0 + nloc] = (bf16_t)(o0[r]*inv);
    Yb[base + 16 + nloc] = (bf16_t)(o1[r]*inv);
    Yb[base + 32 + nloc] = (bf16_t)(o2[r]*inv);
    Yb[base + 48 + nloc] = (bf16_t)(o3[r]*inv);
  }
}

// ---------------------------------------------------------------------------
// Output projection: out[m,n] = sum_k y[m,k]*Wp[n,k] + bp[n], fp32 out.
__global__ void gemm_out(const bf16_t* __restrict__ yb,
                         const bf16_t* __restrict__ wb,
                         const float*  __restrict__ bias,
                         float* __restrict__ out)
{
  const int lane  = threadIdx.x & 31;
  const int wv    = threadIdx.x >> 5;
  const int mtile = (blockIdx.x * 4 + wv) * 16;
  const int n0    = blockIdx.y * 64;
  const int nloc  = lane & 15, hf = lane >> 4;

  const bf16_t* arow = yb + (size_t)(mtile + nloc) * CEMB + hf * 16;
  const bf16_t* bp0  = wb + (size_t)(n0 +  0 + nloc) * CEMB + hf * 16;
  const bf16_t* bp1  = wb + (size_t)(n0 + 16 + nloc) * CEMB + hf * 16;
  const bf16_t* bp2  = wb + (size_t)(n0 + 32 + nloc) * CEMB + hf * 16;
  const bf16_t* bp3  = wb + (size_t)(n0 + 48 + nloc) * CEMB + hf * 16;

  v8f a0 = {}, a1 = {}, a2 = {}, a3 = {};
  #pragma unroll 4
  for (int kb = 0; kb < CEMB; kb += 32){
    v16bf af = ldA(arow + kb);
    a0 = wmma_bf(af, ldB(bp0 + kb), a0);
    a1 = wmma_bf(af, ldB(bp1 + kb), a1);
    a2 = wmma_bf(af, ldB(bp2 + kb), a2);
    a3 = wmma_bf(af, ldB(bp3 + kb), a3);
  }
  #pragma unroll
  for (int j = 0; j < 4; j++){
    v8f acc = (j==0)?a0:(j==1)?a1:(j==2)?a2:a3;
    const int n = n0 + j*16 + nloc;
    const float bv = bias[n];
    #pragma unroll
    for (int r = 0; r < 8; r++){
      const int mg = mtile + r + hf*8;
      out[(size_t)mg*CEMB + n] = acc[r] + bv;
    }
  }
}

// ---------------------------------------------------------------------------
extern "C" void kernel_launch(void* const* d_in, const int* in_sizes, int n_in,
                              void* d_out, int out_size, void* d_ws, size_t ws_size,
                              hipStream_t stream)
{
  (void)in_sizes; (void)n_in; (void)out_size; (void)ws_size;
  const float* x  = (const float*)d_in[0];
  const float* Wq = (const float*)d_in[1];
  const float* bq = (const float*)d_in[2];
  const float* Wk = (const float*)d_in[3];
  const float* bk = (const float*)d_in[4];
  const float* Wv = (const float*)d_in[5];
  const float* bv = (const float*)d_in[6];
  const float* Wp = (const float*)d_in[7];
  const float* bp = (const float*)d_in[8];
  float* out = (float*)d_out;

  char* ws = (char*)d_ws;
  size_t off = 0;
  auto suballoc = [&](size_t bytes) -> char* {
    char* p = ws + off; off += (bytes + 255) & ~(size_t)255; return p;
  };
  const size_t xbytes = (size_t)MTOT * CEMB * sizeof(bf16_t);
  const size_t wbytes = (size_t)CEMB * CEMB * sizeof(bf16_t);
  bf16_t* xb  = (bf16_t*)suballoc(xbytes);
  bf16_t* wqb = (bf16_t*)suballoc(wbytes);
  bf16_t* wkb = (bf16_t*)suballoc(wbytes);
  bf16_t* wvb = (bf16_t*)suballoc(wbytes);
  bf16_t* wpb = (bf16_t*)suballoc(wbytes);
  bf16_t* Qbf = (bf16_t*)suballoc(xbytes);   // [b][h][t][d]
  bf16_t* Kbf = (bf16_t*)suballoc(xbytes);   // [b][h][t][d]
  bf16_t* Vtb = (bf16_t*)suballoc(xbytes);   // [b][h][d][t]
  bf16_t* Ybf = (bf16_t*)suballoc(xbytes);   // [b][t][h*d]

  cvt_bf16<<<2048, 256, 0, stream>>>(x,  xb,  MTOT*CEMB);
  cvt_bf16<<<1024, 256, 0, stream>>>(Wq, wqb, CEMB*CEMB);
  cvt_bf16<<<1024, 256, 0, stream>>>(Wk, wkb, CEMB*CEMB);
  cvt_bf16<<<1024, 256, 0, stream>>>(Wv, wvb, CEMB*CEMB);
  cvt_bf16<<<1024, 256, 0, stream>>>(Wp, wpb, CEMB*CEMB);

  dim3 gg(MTOT/64, CEMB/64);                 // (128, 12), 4 waves/block
  gemm_qkv<<<gg, 128, 0, stream>>>(xb, wqb, bq, Qbf, 0);
  gemm_qkv<<<gg, 128, 0, stream>>>(xb, wkb, bk, Kbf, 1);
  gemm_qkv<<<gg, 128, 0, stream>>>(xb, wvb, bv, Vtb, 2);

  flash_attn<<<dim3(TSEQ/64, NHEAD, NBAT), 128, 0, stream>>>(Qbf, Kbf, Vtb, Ybf);

  gemm_out<<<gg, 128, 0, stream>>>(Ybf, wpb, bp, out);
}
